// SwigluMoEBlock_1967095021959
// MI455X (gfx1250) — compile-verified, matrix operand order checked by reading, not run on hardware
//
#include <hip/hip_runtime.h>
#include <stdint.h>

// Problem constants (match reference setup_inputs)
#define ALPHA_ 1.702f
#define BETA_  1.0f
#define LIMIT_ 7.0f
#define B_  4
#define S_  1024
#define H_  1024
#define E_  8
#define I_  512
#define T_  (B_*S_)           // 4096 tokens
#define XS_        (H_ + 8)   // LDS X row stride (bf16 elems): 2064 B -> bank-spread
#define ACT_STRIDE (I_ + 8)   // LDS act row stride (bf16 elems): 1040 B -> bank-spread

typedef __attribute__((ext_vector_type(16))) __bf16 v16bf;
typedef __attribute__((ext_vector_type(8)))  float  v8f;

union Frag {
    v16bf v;
    uint4 q[2];
};

static __device__ __forceinline__ uint16_t f32_to_bf16(float f) {
    uint32_t u = __float_as_uint(f);
    uint32_t r = u + 0x7FFFu + ((u >> 16) & 1u);   // round-to-nearest-even
    return (uint16_t)(r >> 16);
}

// gfx1250 async global->LDS copy (ASYNCcnt-tracked). VDST = LDS byte offset,
// VADDR = 64-bit global address (GV mode, saddr = off).
static __device__ __forceinline__ void async_ld_b128(uint32_t lds_off, const void* gaddr) {
    asm volatile("global_load_async_to_lds_b128 %0, %1, off"
                 :: "v"(lds_off), "v"(gaddr)
                 : "memory");
}
static __device__ __forceinline__ void wait_asynccnt0() {
    asm volatile("s_wait_asynccnt 0x0" ::: "memory");
}

// ---------------------------------------------------------------------------
// f32 -> bf16 bulk convert, 4-wide
// ---------------------------------------------------------------------------
__global__ void cvt4_kernel(const float4* __restrict__ s, ushort4* __restrict__ d, int n4) {
    int i = blockIdx.x * blockDim.x + threadIdx.x;
    int stride = gridDim.x * blockDim.x;
    for (; i < n4; i += stride) {
        float4 f = s[i];
        ushort4 o;
        o.x = f32_to_bf16(f.x); o.y = f32_to_bf16(f.y);
        o.z = f32_to_bf16(f.z); o.w = f32_to_bf16(f.w);
        d[i] = o;
    }
}

// ---------------------------------------------------------------------------
// Router: one wave (32 lanes) per token. logits -> softmax -> top2 -> renorm,
// append token to per-expert lists.
// ---------------------------------------------------------------------------
__global__ __launch_bounds__(256) void router_kernel(
    const float* __restrict__ x, const float* __restrict__ gw, const float* __restrict__ gb,
    int* __restrict__ cnt, int* __restrict__ list, float* __restrict__ wts)
{
    int lane = threadIdx.x & 31;
    int wave = threadIdx.x >> 5;
    int t = blockIdx.x * 8 + wave;
    if (t >= T_) return;

    const float* xr = x + (size_t)t * H_;
    float acc[E_];
#pragma unroll
    for (int e = 0; e < E_; ++e) acc[e] = 0.f;
    for (int h = lane; h < H_; h += 32) {
        float xv = xr[h];
#pragma unroll
        for (int e = 0; e < E_; ++e) acc[e] = fmaf(xv, gw[e * H_ + h], acc[e]);
    }
#pragma unroll
    for (int e = 0; e < E_; ++e) {
#pragma unroll
        for (int off = 16; off > 0; off >>= 1) acc[e] += __shfl_xor(acc[e], off, 32);
    }
    if (lane == 0) {
        float mx = -1e30f;
#pragma unroll
        for (int e = 0; e < E_; ++e) { acc[e] += gb[e]; mx = fmaxf(mx, acc[e]); }
        float p[E_];
#pragma unroll
        for (int e = 0; e < E_; ++e) p[e] = __expf(acc[e] - mx);  // denom cancels in renorm
        int i0 = 0;
#pragma unroll
        for (int e = 1; e < E_; ++e) if (p[e] > p[i0]) i0 = e;
        int i1 = (i0 == 0) ? 1 : 0;
#pragma unroll
        for (int e = 0; e < E_; ++e) if (e != i0 && p[e] > p[i1]) i1 = e;
        float w0 = p[i0], w1 = p[i1];
        float inv = 1.0f / (w0 + w1);
        w0 *= inv; w1 *= inv;
        int p0 = atomicAdd(&cnt[i0], 1);
        list[i0 * T_ + p0] = t;  wts[i0 * T_ + p0] = w0;
        int p1 = atomicAdd(&cnt[i1], 1);
        list[i1 * T_ + p1] = t;  wts[i1 * T_ + p1] = w1;
    }
}

// ---------------------------------------------------------------------------
// Expert FFN: one block per (expert, 32-token chunk). 256 threads = 8 waves.
// Stage 0: async-copy 32 gathered X rows (bf16) into LDS (ASYNCcnt path).
//          Rows beyond the expert's count are clamped to a valid token; their
//          results are computed but never scattered, so no masking is needed.
// Phase 1: act(32 x I) = swiglu(X@w1^T, X@w3^T) -> LDS (bf16), WMMA bf16.
// Phase 2: y(32 x H)  = act @ w2^T ; scaled atomic f32 scatter into out.
// WMMA fragment layout per CDNA5 ISA (bf16 16x32):
//   lane L: row = L%16 (+tile base); half = L/16
//   chunk0 = 8 bf16 at K = kbase + half*8 ; chunk1 = 8 bf16 at K = kbase+16+half*8
// k-loops are unroll-capped at 4 to stay under 256 VGPRs (no MSB banking).
// ---------------------------------------------------------------------------
__global__ __launch_bounds__(256) void ffn_kernel(
    const uint16_t* __restrict__ Xbf,
    const uint16_t* __restrict__ W1, const float* __restrict__ b1,
    const uint16_t* __restrict__ W3, const float* __restrict__ b3,
    const uint16_t* __restrict__ W2, const float* __restrict__ b2,
    const int* __restrict__ cnt, const int* __restrict__ list,
    const float* __restrict__ wts, float* __restrict__ out)
{
    __shared__ __align__(16) uint16_t sX[32 * XS_];          // 66,048 B
    __shared__ __align__(16) uint16_t sAct[32 * ACT_STRIDE]; // 33,280 B

    int e = blockIdx.y;
    int mbase = blockIdx.x * 32;
    int ne = cnt[e];
    if (mbase >= ne) return;

    int tid  = threadIdx.x;
    int lane = tid & 31;
    int wave = tid >> 5;
    int half = lane >> 4;
    int l16  = lane & 15;

    const uint16_t* W1e = W1 + (size_t)e * I_ * H_;
    const uint16_t* W3e = W3 + (size_t)e * I_ * H_;
    const uint16_t* W2e = W2 + (size_t)e * H_ * I_;
    const int*   liste = list + e * T_;
    const float* wtse  = wts  + e * T_;

    // ---- Stage 0: gather 32 X rows into LDS with async b128 copies ----
    // 32 rows x 128 16B-chunks = 4096 chunks; 16 per thread.
#pragma unroll
    for (int i = 0; i < 16; ++i) {
        int g   = i * 256 + tid;        // chunk id 0..4095
        int row = g >> 7;               // 0..31
        int c16 = g & 127;              // 16B chunk within row
        int rl  = mbase + row;
        int tok = liste[rl < ne ? rl : (ne - 1)];
        const uint16_t* gp = Xbf + (size_t)tok * H_ + c16 * 8;
        uint32_t loff = (uint32_t)(uintptr_t)(&sX[row * XS_ + c16 * 8]);
        async_ld_b128(loff, gp);
    }
    wait_asynccnt0();
    __syncthreads();

    // ---- Phase 1: 2 (M) x 32 (N) tiles of 16x16 over K=H ----
    for (int tile = wave; tile < 64; tile += 8) {
        int msub = tile & 1;
        int nt   = tile >> 1;
        const uint16_t* arow = &sX[(msub * 16 + l16) * XS_];
        int ncol = nt * 16 + l16;
        const uint16_t* b1row = W1e + (size_t)ncol * H_;
        const uint16_t* b3row = W3e + (size_t)ncol * H_;

        v8f accg = {}; v8f accv = {};
#pragma unroll 4
        for (int k = 0; k < H_; k += 32) {
            Frag a, bg, bv;
            const uint16_t* ap = arow + k + half * 8;
            a.q[0]  = *(const uint4*)(ap);
            a.q[1]  = *(const uint4*)(ap + 16);
            const uint16_t* bp1 = b1row + k + half * 8;
            bg.q[0] = *(const uint4*)(bp1);
            bg.q[1] = *(const uint4*)(bp1 + 16);
            const uint16_t* bp3 = b3row + k + half * 8;
            bv.q[0] = *(const uint4*)(bp3);
            bv.q[1] = *(const uint4*)(bp3 + 16);
            accg = __builtin_amdgcn_wmma_f32_16x16x32_bf16(false, a.v, false, bg.v,
                                                           (short)0, accg, false, false);
            accv = __builtin_amdgcn_wmma_f32_16x16x32_bf16(false, a.v, false, bv.v,
                                                           (short)0, accv, false, false);
        }
        float bb1 = b1[e * I_ + ncol];
        float bb3 = b3[e * I_ + ncol];
#pragma unroll
        for (int j = 0; j < 8; ++j) {
            float g  = accg[j] + bb1;
            float vv = accv[j] + bb3;
            g  = fminf(g, LIMIT_);
            vv = fminf(fmaxf(vv, -LIMIT_), LIMIT_);
            float sig = 1.0f / (1.0f + __expf(-ALPHA_ * g));
            float av  = g * sig * (vv + BETA_);
            int rl = msub * 16 + j + half * 8;
            sAct[rl * ACT_STRIDE + ncol] = f32_to_bf16(av);
        }
    }
    __syncthreads();

    // ---- Phase 2: 2 (M) x 64 (N) tiles of 16x16 over K=I ----
    for (int tile = wave; tile < 128; tile += 8) {
        int msub = tile & 1;
        int nt   = tile >> 1;
        const uint16_t* arow = &sAct[(msub * 16 + l16) * ACT_STRIDE];
        int hcol = nt * 16 + l16;
        const uint16_t* brow = W2e + (size_t)hcol * I_;

        v8f acc = {};
#pragma unroll 4
        for (int k = 0; k < I_; k += 32) {
            Frag a, b;
            const uint16_t* ap = arow + k + half * 8;
            a.q[0] = *(const uint4*)(ap);
            a.q[1] = *(const uint4*)(ap + 16);
            const uint16_t* bp = brow + k + half * 8;
            b.q[0] = *(const uint4*)(bp);
            b.q[1] = *(const uint4*)(bp + 16);
            acc = __builtin_amdgcn_wmma_f32_16x16x32_bf16(false, a.v, false, b.v,
                                                          (short)0, acc, false, false);
        }
        float bb2 = b2[e * H_ + hcol];
#pragma unroll
        for (int j = 0; j < 8; ++j) {
            int rl = mbase + msub * 16 + j + half * 8;
            if (rl < ne) {
                int tok = liste[rl];
                float w = wtse[rl];
                atomicAdd(out + (size_t)tok * H_ + hcol, (acc[j] + bb2) * w);
            }
        }
    }
}

// ---------------------------------------------------------------------------
// Host launcher
// ---------------------------------------------------------------------------
extern "C" void kernel_launch(void* const* d_in, const int* in_sizes, int n_in,
                              void* d_out, int out_size, void* d_ws, size_t ws_size,
                              hipStream_t stream) {
    (void)in_sizes; (void)n_in; (void)ws_size;
    const float* x  = (const float*)d_in[0];
    const float* gw = (const float*)d_in[1];
    const float* gb = (const float*)d_in[2];
    const float* w1 = (const float*)d_in[3];
    const float* b1 = (const float*)d_in[4];
    const float* w3 = (const float*)d_in[5];
    const float* b3 = (const float*)d_in[6];
    const float* w2 = (const float*)d_in[7];
    const float* b2 = (const float*)d_in[8];
    // d_in[9] = top_k (compile-time K=2)

    char* ws = (char*)d_ws;
    size_t off = 0;
    auto alloc = [&](size_t bytes) -> char* {
        char* p = ws + off;
        off += (bytes + 255) & ~(size_t)255;
        return p;
    };
    uint16_t* Xbf  = (uint16_t*)alloc((size_t)T_ * H_ * 2);
    uint16_t* W1bf = (uint16_t*)alloc((size_t)E_ * I_ * H_ * 2);
    uint16_t* W3bf = (uint16_t*)alloc((size_t)E_ * I_ * H_ * 2);
    uint16_t* W2bf = (uint16_t*)alloc((size_t)E_ * H_ * I_ * 2);
    int*      cnt  = (int*)  alloc(E_ * sizeof(int));
    int*      list = (int*)  alloc((size_t)E_ * T_ * sizeof(int));
    float*    wts  = (float*)alloc((size_t)E_ * T_ * sizeof(float));

    hipMemsetAsync(cnt, 0, E_ * sizeof(int), stream);
    hipMemsetAsync(d_out, 0, (size_t)out_size * sizeof(float), stream);

    cvt4_kernel<<<1024, 256, 0, stream>>>((const float4*)x,  (ushort4*)Xbf,  (T_ * H_) / 4);
    cvt4_kernel<<<1024, 256, 0, stream>>>((const float4*)w1, (ushort4*)W1bf, (E_ * I_ * H_) / 4);
    cvt4_kernel<<<1024, 256, 0, stream>>>((const float4*)w3, (ushort4*)W3bf, (E_ * I_ * H_) / 4);
    cvt4_kernel<<<1024, 256, 0, stream>>>((const float4*)w2, (ushort4*)W2bf, (E_ * H_ * I_) / 4);

    router_kernel<<<T_ / 8, 256, 0, stream>>>(x, gw, gb, cnt, list, wts);

    dim3 grid(T_ / 32, E_);
    ffn_kernel<<<grid, 256, 0, stream>>>(Xbf, W1bf, b1, W3bf, b3, W2bf, b2,
                                         cnt, list, wts, (float*)d_out);
}
